// FlowMatchingGNN_41644002902519
// MI455X (gfx1250) — matrix operand order repaired and checked by reading, not compiled, not run on hardware
//
#include <hip/hip_runtime.h>

typedef __attribute__((ext_vector_type(16))) _Float16 v16h;
typedef __attribute__((ext_vector_type(8)))  _Float16 v8h;
typedef __attribute__((ext_vector_type(8)))  float    v8f;

// ---------------------------------------------------------------- utilities

__global__ void zero_f32(float* __restrict__ p, long long n) {
    long long t = (long long)blockIdx.x * blockDim.x + threadIdx.x;
    if (t < n) p[t] = 0.0f;
}

// deg[col]++ for every edge (self-loops added later as +1)
__global__ void degree_kernel(const long long* __restrict__ col, float* __restrict__ deg, int E) {
    int t = blockIdx.x * blockDim.x + threadIdx.x;
    if (t < E) atomicAdd(&deg[col[t]], 1.0f);
}

__global__ void dinv_kernel(const float* __restrict__ deg, float* __restrict__ dinv, int n) {
    int t = blockIdx.x * blockDim.x + threadIdx.x;
    if (t < n) dinv[t] = rsqrtf(deg[t] + 1.0f);   // +1 = self-loop; always > 0
}

// -------- pre-shuffle weights into WMMA B-fragment register layout --------
// B fragment layout used here (16-bit, 32x16, wave32, same k order as A): lane holds column n = lane&15;
// element i -> k = (lane>>4)*16 + i.  Stored as B*w[(tile*32 + lane)*16 + i].

// W1 is 16x64, K padded 16->32 with zeros. 4 n-tiles -> 4*32*16 = 2048 halves.
__global__ void prep_B1(const float* __restrict__ W1, _Float16* __restrict__ B1w) {
    int t = blockIdx.x * blockDim.x + threadIdx.x;
    if (t >= 4 * 32 * 16) return;
    int i    = t & 15;
    int lane = (t >> 4) & 31;
    int nt   = t >> 9;
    int k    = (lane >> 4) * 16 + i;
    int n    = nt * 16 + (lane & 15);
    B1w[t] = (k < 16) ? (_Float16)W1[k * 64 + n] : (_Float16)0.0f;
}

// W2 is 64x64: 4 n-tiles x 2 k-chunks -> 8*32*16 = 4096 halves.
__global__ void prep_B2(const float* __restrict__ W2, _Float16* __restrict__ B2w) {
    int t = blockIdx.x * blockDim.x + threadIdx.x;
    if (t >= 8 * 32 * 16) return;
    int i    = t & 15;
    int lane = (t >> 4) & 31;
    int kc   = (t >> 9) & 1;
    int nt   = t >> 10;
    int k    = kc * 32 + (lane >> 4) * 16 + i;
    int n    = nt * 16 + (lane & 15);
    B2w[t] = (_Float16)W2[k * 64 + n];
}

// ------------------------------------------------- GEMM1: xw = x(f32 Nx16) @ W1(16x64), K padded 16->32
// Output buffer is padded to nTiles*16 rows -> stores need no guards.
__global__ void gemm1_wmma(const float* __restrict__ x, const _Float16* __restrict__ B1w,
                           float* __restrict__ xw, int N, int nTiles) {
    int wave = (int)(((long long)blockIdx.x * blockDim.x + threadIdx.x) >> 5);
    int lane = threadIdx.x & 31;
    if (wave >= nTiles) return;                        // wave-uniform: EXEC all-1s at WMMA

    int hh   = lane >> 4;
    int ll   = lane & 15;
    int row0 = wave * 16;
    int m    = row0 + ll;
    int mc   = m < N ? m : N - 1;                      // clamp loads (x has exactly N rows)

    // A: lane holds row mc; element i -> k = hh*16 + i; k>=16 zero-padded
    const float4* xr = (const float4*)(x + (long long)mc * 16);
    float4 xv[4] = {xr[0], xr[1], xr[2], xr[3]};
    v16h a;
#pragma unroll
    for (int i = 0; i < 16; ++i) a[i] = hh ? (_Float16)0.0f : (_Float16)((const float*)xv)[i];

    const v16h* bp = (const v16h*)B1w;

#pragma unroll
    for (int nt = 0; nt < 4; ++nt) {
        v16h b = bp[nt * 32 + lane];                   // coalesced 32B/lane B-fragment
        v8f c = {};
        c = __builtin_amdgcn_wmma_f32_16x16x32_f16(false, a, false, b, (short)0, c, false, false);

        float* op = xw + (long long)(row0 + 8 * hh) * 64 + nt * 16 + ll;
#pragma unroll
        for (int r = 0; r < 8; ++r) op[r * 64] = c[r]; // unguarded: padded buffer
    }
}

// ------------------------------------------------- GEMM2: xw2 = h1(f16 Nx64) @ W2(64x64), K=64 = 2 chunks
// Both h1 and xw2 are padded workspace buffers -> no clamps, no guards.
__global__ void gemm2_wmma(const _Float16* __restrict__ h1, const _Float16* __restrict__ B2w,
                           float* __restrict__ xw2, int nTiles) {
    int wave = (int)(((long long)blockIdx.x * blockDim.x + threadIdx.x) >> 5);
    int lane = threadIdx.x & 31;
    if (wave >= nTiles) return;

    int hh   = lane >> 4;
    int ll   = lane & 15;
    int row0 = wave * 16;
    const _Float16* arow = h1 + (long long)(row0 + ll) * 64;

    // A fragments: element i -> k = kc*32 + hh*16 + i; each 8-run contiguous, 16B aligned
    v16h aA, aB;
    {
        v8h lo  = *(const v8h*)(arow + 0  + hh * 16);
        v8h hi  = *(const v8h*)(arow + 8  + hh * 16);
        v8h lo2 = *(const v8h*)(arow + 32 + hh * 16);
        v8h hi2 = *(const v8h*)(arow + 40 + hh * 16);
#pragma unroll
        for (int i = 0; i < 8; ++i) {
            aA[i] = lo[i];  aA[8 + i] = hi[i];
            aB[i] = lo2[i]; aB[8 + i] = hi2[i];
        }
    }

    const v16h* bp = (const v16h*)B2w;

#pragma unroll
    for (int nt = 0; nt < 4; ++nt) {
        v8f c = {};
        v16h b0 = bp[(nt * 2 + 0) * 32 + lane];
        c = __builtin_amdgcn_wmma_f32_16x16x32_f16(false, aA, false, b0, (short)0, c, false, false);
        v16h b1 = bp[(nt * 2 + 1) * 32 + lane];
        c = __builtin_amdgcn_wmma_f32_16x16x32_f16(false, aB, false, b1, (short)0, c, false, false);

        float* op = xw2 + (long long)(row0 + 8 * hh) * 64 + nt * 16 + ll;
#pragma unroll
        for (int r = 0; r < 8; ++r) op[r * 64] = c[r]; // unguarded: padded buffer
    }
}

// ------------------------------------------------- edge scatter: acc[col] += dinv[row]*dinv[col] * src[row]
// 16 threads per edge, float4 per thread; accumulator is L2-resident (25.6MB << 192MB L2)
__global__ void scatter_edges(const long long* __restrict__ rows, const long long* __restrict__ cols,
                              const float* __restrict__ dinv, const float* __restrict__ src,
                              float* __restrict__ acc, int E) {
    long long t = (long long)blockIdx.x * blockDim.x + threadIdx.x;
    long long e = t >> 4;
    if (e >= E) return;
    int q = (int)(t & 15);
    long long r = rows[e];
    long long c = cols[e];
    float nrm = dinv[r] * dinv[c];
    const float4 v = *(const float4*)(src + r * 64 + q * 4);
    float* dst = acc + c * 64 + q * 4;
    atomicAdd(dst + 0, nrm * v.x);
    atomicAdd(dst + 1, nrm * v.y);
    atomicAdd(dst + 2, nrm * v.z);
    atomicAdd(dst + 3, nrm * v.w);
}

// ------------------------------------------------- finalize layer 1: h1 = relu(acc + dinv^2*xw + b1), store f16
__global__ void finalize1(const float* __restrict__ acc, const float* __restrict__ xw,
                          const float* __restrict__ dinv, const float* __restrict__ b1,
                          _Float16* __restrict__ h1, int total) {
    int t = blockIdx.x * blockDim.x + threadIdx.x;
    if (t >= total) return;
    int i = t >> 6, j = t & 63;
    float di = dinv[i];
    float v = acc[t] + di * di * xw[t] + b1[j];
    h1[t] = (_Float16)fmaxf(v, 0.0f);
}

// ------------------------------------------------- finalize layer 2 (in place): acc = relu(acc + dinv^2*xw2 + b2)
__global__ void finalize2(float* __restrict__ acc, const float* __restrict__ xw2,
                          const float* __restrict__ dinv, const float* __restrict__ b2, int total) {
    int t = blockIdx.x * blockDim.x + threadIdx.x;
    if (t >= total) return;
    int i = t >> 6, j = t & 63;
    float di = dinv[i];
    float v = acc[t] + di * di * xw2[t] + b2[j];
    acc[t] = fmaxf(v, 0.0f);
}

// ------------------------------------------------- head: out = sigmoid(h2 @ Wf + bf)
__global__ void head_kernel(const float* __restrict__ h2, const float* __restrict__ Wf,
                            const float* __restrict__ bf, float* __restrict__ out, int N) {
    int i = blockIdx.x * blockDim.x + threadIdx.x;
    if (i >= N) return;
    float s = bf[0];
    const float4* hp = (const float4*)(h2 + (long long)i * 64);
    const float4* wp = (const float4*)Wf;
#pragma unroll
    for (int q = 0; q < 16; ++q) {
        float4 a = hp[q], w = wp[q];
        s += a.x * w.x + a.y * w.y + a.z * w.z + a.w * w.w;
    }
    out[i] = 1.0f / (1.0f + __expf(-s));
}

// ---------------------------------------------------------------- launcher

extern "C" void kernel_launch(void* const* d_in, const int* in_sizes, int n_in,
                              void* d_out, int out_size, void* d_ws, size_t ws_size,
                              hipStream_t stream) {
    const float*     x   = (const float*)d_in[0];
    const long long* ei  = (const long long*)d_in[1];
    const float*     W1  = (const float*)d_in[2];
    const float*     b1  = (const float*)d_in[3];
    const float*     W2  = (const float*)d_in[4];
    const float*     b2  = (const float*)d_in[5];
    const float*     Wf  = (const float*)d_in[6];
    const float*     bf  = (const float*)d_in[7];
    float*           out = (float*)d_out;

    const int N = in_sizes[0] / 16;
    const int E = in_sizes[1] / 2;
    const long long* rows = ei;
    const long long* cols = ei + E;

    const int nTiles = (N + 15) / 16;
    const int Npad   = nTiles * 16;                    // padded row count for guard-free WMMA stores

    // workspace carve-up (256B aligned)
    char* base = (char*)d_ws;
    size_t off = 0;
    auto carve = [&](size_t bytes) { char* p = base + off; off = (off + bytes + 255) & ~(size_t)255; return p; };
    float*    deg  = (float*)   carve((size_t)N * 4);
    float*    dinv = (float*)   carve((size_t)N * 4);
    float*    bufA = (float*)   carve((size_t)Npad * 64 * 4);   // xw / xw2 (padded)
    float*    bufB = (float*)   carve((size_t)Npad * 64 * 4);   // acc1 / acc2 / h2 (padded)
    _Float16* h1h  = (_Float16*)carve((size_t)Npad * 64 * 2);   // h1 (f16, padded)
    _Float16* B1w  = (_Float16*)carve((size_t)4 * 32 * 16 * 2);
    _Float16* B2w  = (_Float16*)carve((size_t)8 * 32 * 16 * 2);
    (void)ws_size;

    const int B256 = 256;
    const int gemmGrid = (nTiles + 7) / 8;                        // 8 wave32s per 256-thread block
    const long long totNF = (long long)N * 64;
    const int elemGrid  = (int)((totNF + B256 - 1) / B256);
    const long long scatterThreads = (long long)E * 16;
    const int scatterGrid = (int)((scatterThreads + B256 - 1) / B256);

    // weights -> WMMA-layout f16 fragments (tiny)
    prep_B1<<<(4 * 32 * 16 + B256 - 1) / B256, B256, 0, stream>>>(W1, B1w);
    prep_B2<<<(8 * 32 * 16 + B256 - 1) / B256, B256, 0, stream>>>(W2, B2w);

    // degrees + normalization
    zero_f32<<<(N + B256 - 1) / B256, B256, 0, stream>>>(deg, N);
    degree_kernel<<<(E + B256 - 1) / B256, B256, 0, stream>>>(cols, deg, E);
    dinv_kernel<<<(N + B256 - 1) / B256, B256, 0, stream>>>(deg, dinv, N);

    // ---- layer 1
    zero_f32<<<elemGrid, B256, 0, stream>>>(bufB, totNF);
    gemm1_wmma<<<gemmGrid, B256, 0, stream>>>(x, B1w, bufA, N, nTiles);
    scatter_edges<<<scatterGrid, B256, 0, stream>>>(rows, cols, dinv, bufA, bufB, E);
    finalize1<<<elemGrid, B256, 0, stream>>>(bufB, bufA, dinv, b1, h1h, (int)totNF);

    // ---- layer 2
    gemm2_wmma<<<gemmGrid, B256, 0, stream>>>(h1h, B2w, bufA, nTiles);
    zero_f32<<<elemGrid, B256, 0, stream>>>(bufB, totNF);
    scatter_edges<<<scatterGrid, B256, 0, stream>>>(rows, cols, dinv, bufA, bufB, E);
    finalize2<<<elemGrid, B256, 0, stream>>>(bufB, bufA, dinv, b2, (int)totNF);

    // ---- head
    head_kernel<<<(N + B256 - 1) / B256, B256, 0, stream>>>(bufB, Wf, bf, out, N);
}
